// _denoising_block_21002390077568
// MI455X (gfx1250) — matrix operand order repaired and checked by reading, not compiled, not run on hardware
//
#include <hip/hip_runtime.h>
#include <math.h>

// ---------------- problem constants ----------------
#define BB 4
#define CC 64
#define HH 96
#define WW 96
#define HW (HH*WW)          // 9216
#define HS 4
#define S_TOK (HW/HS)       // 2304 tokens per head
#define NCH (S_TOK/32)      // 72 t-chunks of 32

typedef __bf16 bf16_t;
typedef __attribute__((ext_vector_type(16))) __bf16 v16bf;
typedef __attribute__((ext_vector_type(8)))  __bf16 v8bf;
typedef __attribute__((ext_vector_type(8)))  float  v8f;
typedef __attribute__((ext_vector_type(2)))  float  v2f;

// load 16 contiguous bf16 (32B, 16B-aligned) into a v16bf fragment
__device__ __forceinline__ v16bf load16(const bf16_t* p) {
    v8bf a = *(const v8bf*)p;
    v8bf b = *(const v8bf*)(p + 8);
    v16bf r;
#pragma unroll
    for (int j = 0; j < 8; ++j) { r[j] = a[j]; r[j + 8] = b[j]; }
    return r;
}

__device__ __forceinline__ v16bf pack16(v8bf lo, v8bf hi) {
    v16bf r;
#pragma unroll
    for (int j = 0; j < 8; ++j) { r[j] = lo[j]; r[j + 8] = hi[j]; }
    return r;
}

// DPP16 ROW_XMASK xor-shuffle within 16-lane rows (pure VALU, no LDS)
#define DPP_XMASK(x, m) \
    __int_as_float(__builtin_amdgcn_update_dpp(__float_as_int(x), __float_as_int(x), \
                                               0x160 | (m), 0xF, 0xF, false))
#define ASYNC_WAIT(n) asm volatile("s_wait_asynccnt " #n ::: "memory")

// ---------------- kernel 0: weights fp32 -> bf16 ----------------
__global__ __launch_bounds__(256) void conv_w(
    const float* __restrict__ Wq, const float* __restrict__ Wk, const float* __restrict__ Wv,
    bf16_t* __restrict__ Wqb, bf16_t* __restrict__ Wkb, bf16_t* __restrict__ Wvb)
{
    const int i = blockIdx.x * 256 + threadIdx.x;   // 4096 per matrix
    Wqb[i] = (bf16_t)Wq[i];
    Wkb[i] = (bf16_t)Wk[i];
    Wvb[i] = (bf16_t)Wv[i];
}

// ---------------- kernel 1: QKV projection via WMMA (bf16) ----------------
__global__ __launch_bounds__(256) void qkv_wmma(
    const float* __restrict__ x,
    const bf16_t* __restrict__ Wqb, const bf16_t* __restrict__ Wkb, const bf16_t* __restrict__ Wvb,
    const float* __restrict__ bq, const float* __restrict__ bk, const float* __restrict__ bv,
    bf16_t* __restrict__ Qt, bf16_t* __restrict__ Kt, bf16_t* __restrict__ Vc)
{
    __shared__ __align__(16) bf16_t xt[128][72];    // 128 tokens x 64 c (+8 pad)

    const int tidb = threadIdx.x;
    const int wave = tidb >> 5, lane = tidb & 31;
    const int half = lane >> 4,  n    = lane & 15;
    const int b    = blockIdx.x / (HW / 128);
    const int hw0  = (blockIdx.x % (HW / 128)) * 128;

    // coalesced read of x[64c][128hw], transposed bf16 store to LDS
    {
        const int c0  = tidb >> 7;
        const int hwl = tidb & 127;
#pragma unroll
        for (int k = 0; k < 32; ++k) {
            const int c = k * 2 + c0;
            xt[hwl][c] = (bf16_t)x[((long)b * CC + c) * HW + hw0 + hwl];
        }
    }
    __syncthreads();

    const int t0l = wave * 16;
    v16bf xa[2];
#pragma unroll
    for (int ch = 0; ch < 2; ++ch) {
        const bf16_t* row = &xt[t0l + n][ch * 32 + half * 16];
        xa[ch] = pack16(*(const v8bf*)row, *(const v8bf*)(row + 8));
    }

    const v8f zero = {0, 0, 0, 0, 0, 0, 0, 0};
    const int tokbase = b * HW + hw0 + t0l;
    const int bhv     = tokbase / S_TOK;
    const int sv      = tokbase - bhv * S_TOK;

#pragma unroll
    for (int p = 0; p < 3; ++p) {
        const bf16_t* W    = (p == 0) ? Wqb : (p == 1) ? Wkb : Wvb;
        const float* bias  = (p == 0) ? bq  : (p == 1) ? bk  : bv;
        const float  bmul  = (p == 0) ? 0.5f : 1.0f;      // fold softmax scale into Q
#pragma unroll
        for (int ot = 0; ot < 4; ++ot) {
            const int o = ot * 16 + n;
            const v16bf wb0 = load16(&W[o * CC + half * 16]);
            const v16bf wb1 = load16(&W[o * CC + 32 + half * 16]);
            v8f d = __builtin_amdgcn_wmma_f32_16x16x32_bf16(false, xa[0], false, wb0,
                                                            (short)0, zero, false, false);
            d = __builtin_amdgcn_wmma_f32_16x16x32_bf16(false, xa[1], false, wb1,
                                                        (short)0, d, false, false);
            const float bo_ = bias[o];
            if (p < 2) {                            // Q/K token-major (tok, c=o)
                bf16_t* dst = (p == 0 ? Qt : Kt) + (long)(tokbase + half * 8) * CC + o;
#pragma unroll
                for (int r = 0; r < 8; ++r)
                    dst[(long)r * CC] = (bf16_t)((d[r] + bo_) * bmul);
            } else {                                // V channel-major (c=o, s)
                bf16_t* dst = Vc + ((long)bhv * CC + o) * S_TOK + sv + half * 8;
#pragma unroll
                for (int r = 0; r < 8; ++r)         // 8 contiguous bf16 -> b128 store
                    dst[r] = (bf16_t)(d[r] + bo_);
            }
        }
    }
}

// ---------------- kernel 2: flash attention, bf16 WMMA + async-LDS K/V ----------------
__global__ __launch_bounds__(256) void attn_fa(
    const bf16_t* __restrict__ Qt, const bf16_t* __restrict__ Kt,
    const bf16_t* __restrict__ Vc, float* __restrict__ O)
{
    __shared__ __align__(16) bf16_t kbuf[3][32][72];
    __shared__ __align__(16) bf16_t vbuf[3][64][40];
    __shared__ __align__(16) bf16_t plds[8][16][32];

    const int tidb = threadIdx.x;
    const int wave = tidb >> 5;
    const int lane = tidb & 31;
    const int half = lane >> 4;
    const int n    = lane & 15;
    const int bh   = blockIdx.y;
    const int s0   = (blockIdx.x * 8 + wave) * 16;

    const bf16_t* qbase = Qt + (long)bh * S_TOK * CC;
    const bf16_t* kbase = Kt + (long)bh * S_TOK * CC;
    const bf16_t* vbase = Vc + (long)bh * CC * S_TOK;

    auto issueKV = [&](int i) {
        const int bi = i % 3;
        const int t0 = i * 32;
        const unsigned kl   = (unsigned)(uintptr_t)&kbuf[bi][tidb >> 3][(tidb & 7) * 8];
        const unsigned koff = (unsigned)(((t0 + (tidb >> 3)) * CC + (tidb & 7) * 8) * 2);
        asm volatile("global_load_async_to_lds_b128 %0, %1, %2"
                     :: "v"(kl), "v"(koff), "s"(kbase) : "memory");
        const unsigned vl   = (unsigned)(uintptr_t)&vbuf[bi][tidb >> 2][(tidb & 3) * 8];
        const unsigned voff = (unsigned)(((tidb >> 2) * S_TOK + t0 + (tidb & 3) * 8) * 2);
        asm volatile("global_load_async_to_lds_b128 %0, %1, %2"
                     :: "v"(vl), "v"(voff), "s"(vbase) : "memory");
    };

    v16bf qa[2];
#pragma unroll
    for (int ch = 0; ch < 2; ++ch) {
        const bf16_t* qrow = qbase + (long)(s0 + n) * CC + ch * 32 + half * 16;
        qa[ch] = pack16(*(const v8bf*)qrow, *(const v8bf*)(qrow + 8));
    }

    float mrow[8], lrow[8];
    v8f acc[4];
#pragma unroll
    for (int r = 0; r < 8; ++r) { mrow[r] = -INFINITY; lrow[r] = 0.0f; }
#pragma unroll
    for (int cc = 0; cc < 4; ++cc)
#pragma unroll
        for (int r = 0; r < 8; ++r) acc[cc][r] = 0.0f;

    const v8f zero = {0, 0, 0, 0, 0, 0, 0, 0};
    bf16_t (*pt)[32] = plds[wave];

    issueKV(0);
    issueKV(1);

    for (int i = 0; i < NCH; ++i) {
        if (i == NCH - 1) { ASYNC_WAIT(0); } else { ASYNC_WAIT(2); }
        __syncthreads();
        const int bi = i % 3;

        v16bf kf[4], vf[4];
        kf[0] = load16(&kbuf[bi][n][half * 16]);
        kf[1] = load16(&kbuf[bi][n][32 + half * 16]);
        kf[2] = load16(&kbuf[bi][16 + n][half * 16]);
        kf[3] = load16(&kbuf[bi][16 + n][32 + half * 16]);
#pragma unroll
        for (int cc = 0; cc < 4; ++cc) vf[cc] = load16(&vbuf[bi][cc * 16 + n][half * 16]);

        v8f l0 = zero, l1 = zero;
        l0 = __builtin_amdgcn_wmma_f32_16x16x32_bf16(false, qa[0], false, kf[0], (short)0, l0, false, false);
        l0 = __builtin_amdgcn_wmma_f32_16x16x32_bf16(false, qa[1], false, kf[1], (short)0, l0, false, false);
        l1 = __builtin_amdgcn_wmma_f32_16x16x32_bf16(false, qa[0], false, kf[2], (short)0, l1, false, false);
        l1 = __builtin_amdgcn_wmma_f32_16x16x32_bf16(false, qa[1], false, kf[3], (short)0, l1, false, false);

        float alpha[8];
#pragma unroll
        for (int r = 0; r < 8; ++r) {
            float v = fmaxf(l0[r], l1[r]);
            v = fmaxf(v, DPP_XMASK(v, 1));
            v = fmaxf(v, DPP_XMASK(v, 2));
            v = fmaxf(v, DPP_XMASK(v, 4));
            v = fmaxf(v, DPP_XMASK(v, 8));
            const float mnew = fmaxf(mrow[r], v);
            alpha[r] = __expf(mrow[r] - mnew);
            mrow[r]  = mnew;
            const float p0 = __expf(l0[r] - mnew);
            const float p1 = __expf(l1[r] - mnew);
            l0[r] = p0; l1[r] = p1;
            float ssum = p0 + p1;
            ssum += DPP_XMASK(ssum, 1);
            ssum += DPP_XMASK(ssum, 2);
            ssum += DPP_XMASK(ssum, 4);
            ssum += DPP_XMASK(ssum, 8);
            lrow[r] = lrow[r] * alpha[r] + ssum;
        }

#pragma unroll
        for (int r = 0; r < 8; ++r) {
            const int row = half * 8 + r;
            pt[row][n]      = (bf16_t)l0[r];
            pt[row][n + 16] = (bf16_t)l1[r];
#pragma unroll
            for (int cc = 0; cc < 4; ++cc) acc[cc][r] = acc[cc][r] * alpha[r];
        }

        const bf16_t* prow = &pt[n][0];
        const v16bf pa = pack16(*(const v8bf*)(prow + half * 16),
                                *(const v8bf*)(prow + half * 16 + 8));

#pragma unroll
        for (int cc = 0; cc < 4; ++cc)
            acc[cc] = __builtin_amdgcn_wmma_f32_16x16x32_bf16(false, pa, false, vf[cc],
                                                              (short)0, acc[cc], false, false);

        if (i + 2 < NCH) issueKV(i + 2);
    }

    float* obase = O + (long)bh * CC * S_TOK;
#pragma unroll
    for (int cc = 0; cc < 4; ++cc) {
        const int c = cc * 16 + n;
#pragma unroll
        for (int r = 0; r < 8; ++r) {
            const int srow = s0 + half * 8 + r;
            obase[(long)c * S_TOK + srow] = acc[cc][r] / lrow[r];
        }
    }
}

// ---------------- kernel 3: output projection + residual via fp32 WMMA ----------------
// D(token,o) = O^T(token,c) * Wo^T(c,o) with V_WMMA_F32_16X16X4_F32 (fp32 in,
// fp32 accum -> same precision class as VALU FMA). Result is transposed through
// LDS so the y = x + D + bo epilogue stores fully coalesced 512B/wave.
__global__ __launch_bounds__(256) void out_wmma(
    const float* __restrict__ x, const float* __restrict__ Wo,
    const float* __restrict__ bo, const float* __restrict__ O,
    float* __restrict__ y)
{
    __shared__ __align__(16) float ylds[CC][132];   // 64 o-rows x 128 tokens (+4 pad)

    const int tidb = threadIdx.x;
    const int wave = tidb >> 5, lane = tidb & 31;
    const int half = lane >> 4, n = lane & 15;
    const int b    = blockIdx.x / (HW / 128);
    const int hw0  = (blockIdx.x % (HW / 128)) * 128;
    const int t0l  = wave * 16;

    // A-fragments: O^T (M=token=n, K = 4k + 2*half + j); coalesced 64B segments
    v2f af[16];
    {
        const long base = (long)b * CC * HW + hw0 + t0l + n;
#pragma unroll
        for (int k = 0; k < 16; ++k) {
            const int c0 = 4 * k + half * 2;
            af[k][0] = O[base + (long)(c0    ) * HW];
            af[k][1] = O[base + (long)(c0 + 1) * HW];
        }
    }

    const v8f zero = {0, 0, 0, 0, 0, 0, 0, 0};
#pragma unroll
    for (int ot = 0; ot < 4; ++ot) {
        const int o = ot * 16 + n;
        v8f acc = zero;
#pragma unroll
        for (int k = 0; k < 16; ++k) {
            // B-fragment: Wo^T (K = 4k + 2*half + j, N = o) -> 2 contiguous floats
            const v2f wb = *(const v2f*)&Wo[o * CC + 4 * k + half * 2];
            acc = __builtin_amdgcn_wmma_f32_16x16x4_f32(false, af[k], false, wb,
                                                        (short)0, acc, false, false);
        }
#pragma unroll
        for (int r = 0; r < 8; ++r)
            ylds[o][t0l + half * 8 + r] = acc[r];
    }
    __syncthreads();

    // cooperative epilogue: per wave, 32 lanes sweep one o-row (512B) at a time
    const int seg = lane;                 // 16B segment within the 128-token row
#pragma unroll
    for (int it = 0; it < 8; ++it) {
        const int o = it * 8 + wave;
        const float bo_ = bo[o];
        const long gbase = ((long)b * CC + o) * HW + hw0 + seg * 4;
        const float4 xv = *(const float4*)&x[gbase];
        float4 yv;
        yv.x = xv.x + ylds[o][seg * 4 + 0] + bo_;
        yv.y = xv.y + ylds[o][seg * 4 + 1] + bo_;
        yv.z = xv.z + ylds[o][seg * 4 + 2] + bo_;
        yv.w = xv.w + ylds[o][seg * 4 + 3] + bo_;
        *(float4*)&y[gbase] = yv;
    }
}

// ---------------- launch ----------------
extern "C" void kernel_launch(void* const* d_in, const int* in_sizes, int n_in,
                              void* d_out, int out_size, void* d_ws, size_t ws_size,
                              hipStream_t stream) {
    (void)in_sizes; (void)n_in; (void)out_size; (void)ws_size;
    // setup_inputs order: x, Wq, Wk, Wv, Wo, bq, bk, bv, bo
    const float* x  = (const float*)d_in[0];
    const float* Wq = (const float*)d_in[1];
    const float* Wk = (const float*)d_in[2];
    const float* Wv = (const float*)d_in[3];
    const float* Wo = (const float*)d_in[4];
    const float* bq = (const float*)d_in[5];
    const float* bk = (const float*)d_in[6];
    const float* bv = (const float*)d_in[7];
    const float* bo = (const float*)d_in[8];
    float* y = (float*)d_out;

    const size_t qelems = (size_t)BB * HS * S_TOK * CC;      // 2,359,296
    bf16_t* Qt  = (bf16_t*)d_ws;
    bf16_t* Kt  = Qt + qelems;
    bf16_t* Vc  = Kt + qelems;
    float*  O   = (float*)(Vc + qelems);
    bf16_t* Wqb = (bf16_t*)(O + qelems);                     // 3 x 4096 bf16
    bf16_t* Wkb = Wqb + CC * CC;
    bf16_t* Wvb = Wkb + CC * CC;

    conv_w<<<dim3((CC * CC) / 256), dim3(256), 0, stream>>>(Wq, Wk, Wv, Wqb, Wkb, Wvb);

    qkv_wmma<<<dim3((BB * HW) / 128), dim3(256), 0, stream>>>(
        x, Wqb, Wkb, Wvb, bq, bk, bv, Qt, Kt, Vc);

    attn_fa<<<dim3(S_TOK / 128, BB * HS), dim3(256), 0, stream>>>(Qt, Kt, Vc, O);

    out_wmma<<<dim3((BB * HW) / 128), dim3(256), 0, stream>>>(x, Wo, bo, O, y);
}
